// VehicleDetectionModel_8504035246678
// MI455X (gfx1250) — compile-verified, matrix operand order checked by reading, not run on hardware
//
#include <hip/hip_runtime.h>
#include <hip/hip_bf16.h>
#include <stdint.h>

// ---------------------------------------------------------------------------
// CDNA5 (gfx1250) vector types for WMMA / TDM operands
// ---------------------------------------------------------------------------
typedef __attribute__((ext_vector_type(2)))  float        v2f;
typedef __attribute__((ext_vector_type(8)))  float        v8f;
typedef __attribute__((ext_vector_type(16))) __bf16       v16bf;
typedef __attribute__((ext_vector_type(4)))  unsigned int u32x4;
typedef __attribute__((ext_vector_type(8)))  int          i32x8;
typedef __attribute__((ext_vector_type(4)))  int          i32x4;

// ---------------------------------------------------------------------------
// Problem constants (match reference)
// ---------------------------------------------------------------------------
#define BATCH   2
#define CIN     960
#define FH      20
#define FW      40
#define NPOS    800          // FH*FW
#define NANCH   12
#define NTOT    9600         // NPOS*NANCH
#define PRE_K   1000
#define POST_K  500
#define POOLSZ  7
#define FCK     47040        // CIN*POOLSZ*POOLSZ
#define HID     1024
#define STRIDE_F 16.0f
#define NMS_THR  0.7f
#define BBOX_CLIP_C 4.1351665567f   // log(1000/16)
#define KSLAB   128          // bf16 K elements staged per TDM slab

__device__ inline int imin(int a, int b) { return a < b ? a : b; }

// f32 -> bf16 round-to-nearest-even
__device__ inline uint16_t f2bf(float f) {
  uint32_t x = __float_as_uint(f);
  uint32_t r = x + 0x7FFFu + ((x >> 16) & 1u);
  if ((x & 0x7FFFFFFFu) > 0x7F800000u) r = x | 0x00400000u;  // quiet NaN
  return (uint16_t)(r >> 16);
}

// ---------------------------------------------------------------------------
// Tensor Data Mover: DMA a 2-D tile (tile_rows x tile_k bf16, row stride
// row_stride elements) from global memory into LDS at lds_addr, packed
// contiguously row-major (pitch = tile_k elements).  D# layout per
// CDNA5 ISA (group0: count/lds/global/type=2;
// group1: data_size=2B, tensor dims == tile dims -> no OOB clip, dim0 stride).
// 6-arg builtin form (this toolchain): groups 2/3 + extra group zeroed (2-D).
// ---------------------------------------------------------------------------
__device__ inline void tdm_load_2d(uint32_t lds_addr, const uint16_t* gaddr,
                                   int tile_k, int tile_rows, int row_stride)
{
  const uint64_t ga = (uint64_t)(uintptr_t)gaddr;
  u32x4 g0 = {0u, 0u, 0u, 0u};
  g0[0] = 1u;                                            // count=1 (valid user D#)
  g0[1] = lds_addr;                                      // LDS byte address
  g0[2] = (uint32_t)ga;                                  // global_addr[31:0]
  g0[3] = (uint32_t)((ga >> 32) & 0x01FFFFFFu) | (2u << 30);  // addr[56:32] | type=2
  const uint32_t d0 = (uint32_t)tile_k;                  // tensor_dim0 (== tile_dim0)
  const uint32_t d1 = (uint32_t)tile_rows;               // tensor_dim1 (== tile_dim1)
  const uint64_t s0 = (uint64_t)(uint32_t)row_stride;    // tensor_dim0_stride (elems)
  i32x8 g1 = {0, 0, 0, 0, 0, 0, 0, 0};
  g1[0] = (int)(1u << 16);                               // data_size=1 -> 2 bytes
  g1[1] = (int)((d0 & 0xFFFFu) << 16);                   // dim0[15:0]  @ bits 63:48
  g1[2] = (int)((d0 >> 16) | ((d1 & 0xFFFFu) << 16));    // dim0[31:16], dim1[15:0]
  g1[3] = (int)((d1 >> 16) | (d0 << 16));                // dim1[31:16], tile_dim0
  g1[4] = (int)(d1 & 0xFFFFu);                           // tile_dim1 (tile_dim2=0)
  g1[5] = (int)(uint32_t)(s0 & 0xFFFFFFFFu);             // stride0[31:0]
  g1[6] = (int)(uint32_t)((s0 >> 32) & 0xFFFFu);         // stride0[47:32], stride1=0
  g1[7] = 0;
  const i32x4 gz4 = {0, 0, 0, 0};
  const i32x8 gz8 = {0, 0, 0, 0, 0, 0, 0, 0};
  __builtin_amdgcn_tensor_load_to_lds(g0, g1, gz4, gz4, gz8, 0);
}

// ---------------------------------------------------------------------------
// 1) RPN 3x3 conv (SAME, 960->960) as implicit GEMM with V_WMMA_F32_16X16X4_F32
//    (fp32 WMMA keeps score ordering bit-faithful for top-k / NMS).
// ---------------------------------------------------------------------------
__global__ __launch_bounds__(32)
void rpn_conv_wmma(const float* __restrict__ feat, const float* __restrict__ w,
                   const float* __restrict__ bias, float* __restrict__ t)
{
  const int lane = threadIdx.x;
  const int lo = lane & 15, hi = lane >> 4;
  const int n0 = blockIdx.x * 16;     // out-channel tile
  const int m0 = blockIdx.y * 16;     // spatial tile

  const int mA   = m0 + lo;
  const int bA   = mA / NPOS;
  const int posA = mA - bA * NPOS;
  const int yA   = posA / FW, xA = posA - yA * FW;
  const int nB = n0 + lo;

  v8f acc = {0.f,0.f,0.f,0.f,0.f,0.f,0.f,0.f};

  for (int tap = 0; tap < 9; ++tap) {
    const int ky = tap / 3, kx = tap - ky * 3;
    const int yy = yA + ky - 1, xx = xA + kx - 1;
    const bool valid = (yy >= 0) & (yy < FH) & (xx >= 0) & (xx < FW);
    const float* abase = feat + (size_t)bA * CIN * NPOS + yy * FW + xx; // +i*NPOS
    const float* bbase = w + (size_t)nB * (CIN * 9) + tap;              // +i*9
    for (int it = 0; it < CIN / 4; ++it) {
      const int i0 = it * 4 + hi * 2;
      v2f a, b;
      a.x = valid ? abase[(size_t)i0 * NPOS]       : 0.0f;
      a.y = valid ? abase[(size_t)(i0 + 1) * NPOS] : 0.0f;
      b.x = bbase[(size_t)i0 * 9];
      b.y = bbase[(size_t)(i0 + 1) * 9];
      acc = __builtin_amdgcn_wmma_f32_16x16x4_f32(false, a, false, b,
                                                  (short)0, acc, false, false);
    }
  }

  const int nOut = n0 + lo;
  const float bv = bias[nOut];
#pragma unroll
  for (int j = 0; j < 8; ++j) {
    const int m   = m0 + hi * 8 + j;
    const int b   = m / NPOS;
    const int pos = m - b * NPOS;
    float v = acc[j] + bv;
    v = v > 0.0f ? v : 0.0f;          // ReLU
    t[((size_t)b * CIN + nOut) * NPOS + pos] = v;
  }
}

// ---------------------------------------------------------------------------
// 2) RPN cls/reg heads: 60 dot products of length 960 per spatial position.
// ---------------------------------------------------------------------------
__global__ void rpn_head_kernel(const float* __restrict__ t,
                                const float* __restrict__ cls_w, const float* __restrict__ cls_b,
                                const float* __restrict__ reg_w, const float* __restrict__ reg_b,
                                float* __restrict__ obj, float* __restrict__ reg)
{
  const int gid = blockIdx.x * blockDim.x + threadIdx.x;
  const int TOT = BATCH * NPOS * 60;
  if (gid >= TOT) return;
  const int u   = gid % 60;
  const int pos = (gid / 60) % NPOS;
  const int b   = gid / (60 * NPOS);
  const float* T  = t + (size_t)b * CIN * NPOS + pos;
  const float* Wr = (u < 12) ? (cls_w + (size_t)u * CIN)
                             : (reg_w + (size_t)(u - 12) * CIN);
  float s = (u < 12) ? cls_b[u] : reg_b[u - 12];
  for (int c = 0; c < CIN; ++c) s += T[(size_t)c * NPOS] * Wr[c];
  if (u < 12) {
    obj[(size_t)b * NTOT + pos * NANCH + u] = s;
  } else {
    const int r = u - 12, a = r >> 2, comp = r & 3;
    reg[((size_t)b * NTOT + pos * NANCH + a) * 4 + comp] = s;
  }
}

// ---------------------------------------------------------------------------
// 3) top-1000 of 9600 via O(N^2) rank counting (stable, matches lax.top_k)
// ---------------------------------------------------------------------------
__global__ void topk_obj_kernel(const float* __restrict__ obj,
                                float* __restrict__ sel_s, int* __restrict__ sel_i)
{
  const int b = blockIdx.y;
  const int i = blockIdx.x * blockDim.x + threadIdx.x;
  if (i >= NTOT) return;
  const float* S = obj + (size_t)b * NTOT;
  const float s = S[i];
  int rank = 0;
  for (int j = 0; j < NTOT; ++j) {
    const float sj = S[j];
    rank += (sj > s) || ((sj == s) && (j < i));
  }
  if (rank < PRE_K) {
    sel_s[(size_t)b * PRE_K + rank] = s;
    sel_i[(size_t)b * PRE_K + rank] = i;
  }
}

// ---------------------------------------------------------------------------
// 4) anchor generation (on the fly) + box decode + clip
// ---------------------------------------------------------------------------
__global__ void decode_kernel(const int* __restrict__ sel_i,
                              const float* __restrict__ reg,
                              float* __restrict__ boxes)
{
  const int tix = blockIdx.x * blockDim.x + threadIdx.x;
  if (tix >= BATCH * PRE_K) return;
  const int b = tix / PRE_K;
  const int i = sel_i[tix];
  const int a = i % NANCH, pos = i / NANCH;
  const int x = pos % FW, y = pos / FW;
  const float sizes[4]  = {32.f, 64.f, 128.f, 256.f};
  const float ratios[3] = {0.5f, 1.f, 2.f};
  const float hr = sqrtf(ratios[a >> 2]);
  const float wr = 1.0f / hr;
  const float wa = wr * sizes[a & 3], ha = hr * sizes[a & 3];
  const float cxa = x * STRIDE_F, cya = y * STRIDE_F;
  const float* d = reg + ((size_t)b * NTOT + i) * 4;
  const float dx = d[0], dy = d[1];
  const float dw = fminf(d[2], BBOX_CLIP_C), dh = fminf(d[3], BBOX_CLIP_C);
  const float cx = dx * wa + cxa, cy = dy * ha + cya;
  const float w  = expf(dw) * wa,  h = expf(dh) * ha;
  float x1 = cx - 0.5f * w, y1 = cy - 0.5f * h;
  float x2 = cx + 0.5f * w, y2 = cy + 0.5f * h;
  const float Wimg = FW * STRIDE_F, Himg = FH * STRIDE_F;
  x1 = fminf(fmaxf(x1, 0.f), Wimg); x2 = fminf(fmaxf(x2, 0.f), Wimg);
  y1 = fminf(fmaxf(y1, 0.f), Himg); y2 = fminf(fmaxf(y2, 0.f), Himg);
  float* B = boxes + (size_t)tix * 4;
  B[0] = x1; B[1] = y1; B[2] = x2; B[3] = y2;
}

// ---------------------------------------------------------------------------
// 5) greedy NMS (sequential over 1000, boxes + mask LDS-resident) fused with
//    top-500 rank selection -> props.  One workgroup per image.
// ---------------------------------------------------------------------------
__global__ __launch_bounds__(1024)
void nms_select_kernel(const float* __restrict__ boxes,
                       const float* __restrict__ sel_s,
                       float* __restrict__ props)
{
  __shared__ float bx[PRE_K * 4];
  __shared__ float area[PRE_K];
  __shared__ int   mk[PRE_K];
  __shared__ float scm[PRE_K];
  const int b   = blockIdx.x;
  const int tid = threadIdx.x;
  const float* B0 = boxes + (size_t)b * PRE_K * 4;

  for (int i = tid; i < PRE_K; i += blockDim.x) {
    const float x1 = B0[i*4+0], y1 = B0[i*4+1], x2 = B0[i*4+2], y2 = B0[i*4+3];
    bx[i*4+0]=x1; bx[i*4+1]=y1; bx[i*4+2]=x2; bx[i*4+3]=y2;
    area[i] = (x2 - x1) * (y2 - y1);
    mk[i] = 1;
  }
  __syncthreads();

  for (int i = 0; i < PRE_K - 1; ++i) {
    if (mk[i]) {
      const float ax1 = bx[i*4], ay1 = bx[i*4+1], ax2 = bx[i*4+2], ay2 = bx[i*4+3];
      const float aa = area[i];
      for (int j = i + 1 + tid; j < PRE_K; j += blockDim.x) {
        if (!mk[j]) continue;
        const float lx = fmaxf(ax1, bx[j*4]);
        const float ly = fmaxf(ay1, bx[j*4+1]);
        const float rx = fminf(ax2, bx[j*4+2]);
        const float ry = fminf(ay2, bx[j*4+3]);
        const float iw = fmaxf(rx - lx, 0.f), ih = fmaxf(ry - ly, 0.f);
        const float inter = iw * ih;
        const float iou = inter / (aa + area[j] - inter + 1e-6f);
        if (iou > NMS_THR) mk[j] = 0;
      }
    }
    __syncthreads();
  }

  for (int i = tid; i < PRE_K; i += blockDim.x)
    scm[i] = mk[i] ? sel_s[(size_t)b * PRE_K + i] : -__builtin_inff();
  __syncthreads();

  for (int i = tid; i < PRE_K; i += blockDim.x) {
    const float s = scm[i];
    int rank = 0;
    for (int j = 0; j < PRE_K; ++j) {
      const float sj = scm[j];
      rank += (sj > s) || ((sj == s) && (j < i));
    }
    if (rank < POST_K) {
      float* P = props + ((size_t)b * POST_K + rank) * 4;
      P[0] = bx[i*4]; P[1] = bx[i*4+1]; P[2] = bx[i*4+2]; P[3] = bx[i*4+3];
    }
  }
}

// ---------------------------------------------------------------------------
// 6) ROI-align (7x7, 2x2 subsample mean) -> pooled activations in bf16,
//    laid out row-major (roi, c*49 + py*7 + px): the fc6 GEMM A matrix.
// ---------------------------------------------------------------------------
__global__ __launch_bounds__(256)
void roi_align_kernel(const float* __restrict__ feat,
                      const float* __restrict__ props,
                      uint16_t* __restrict__ pooled)
{
  const int c = blockIdx.x * blockDim.x + threadIdx.x;
  if (c >= CIN) return;
  const int cell = blockIdx.y;                // n*49 + py*7 + px
  const int n = cell / 49, pc = cell % 49;
  const int py = pc / POOLSZ, px = pc % POOLSZ;
  const int b = blockIdx.z;

  const float* P = props + ((size_t)b * POST_K + n) * 4;
  const float x1 = P[0] / STRIDE_F, y1 = P[1] / STRIDE_F;
  const float x2 = P[2] / STRIDE_F, y2 = P[3] / STRIDE_F;
  const float rw = fmaxf(x2 - x1, 1.0f), rh = fmaxf(y2 - y1, 1.0f);
  const float bw = rw / (float)POOLSZ, bh = rh / (float)POOLSZ;

  const float* F = feat + ((size_t)b * CIN + c) * NPOS;
  float acc = 0.f;
#pragma unroll
  for (int sy = 0; sy < 2; ++sy) {
    float yy = y1 + ((float)py + (sy + 0.5f) * 0.5f) * bh;
    yy = fminf(fmaxf(yy, 0.f), (float)(FH - 1));
    const int y0 = (int)floorf(yy);
    const int y1i = (y0 + 1 < FH) ? y0 + 1 : FH - 1;
    const float wy = yy - (float)y0;
#pragma unroll
    for (int sx = 0; sx < 2; ++sx) {
      float xx = x1 + ((float)px + (sx + 0.5f) * 0.5f) * bw;
      xx = fminf(fmaxf(xx, 0.f), (float)(FW - 1));
      const int x0 = (int)floorf(xx);
      const int x1i = (x0 + 1 < FW) ? x0 + 1 : FW - 1;
      const float wx = xx - (float)x0;
      acc += F[y0 * FW + x0]  * (1 - wy) * (1 - wx)
           + F[y0 * FW + x1i] * (1 - wy) * wx
           + F[y1i * FW + x0] * wy * (1 - wx)
           + F[y1i * FW + x1i]* wy * wx;
    }
  }
  acc *= 0.25f;
  pooled[(size_t)(b * POST_K + n) * FCK + c * 49 + pc] = f2bf(acc);
}

// ---------------------------------------------------------------------------
// 7) weight repack: fp32 (K,N) row-major -> bf16 (N,K) row-major, so every
//    WMMA B fragment is a run of packed (k,k+1) bf16 pairs.
// ---------------------------------------------------------------------------
__global__ void repack_w_bf16(const float* __restrict__ in,
                              uint16_t* __restrict__ out, int K, int N)
{
  const size_t idx = (size_t)blockIdx.x * blockDim.x + threadIdx.x;
  if (idx >= (size_t)K * N) return;
  const int n = (int)(idx % N);
  const int k = (int)(idx / N);
  out[(size_t)n * K + k] = f2bf(in[idx]);
}

// ---------------------------------------------------------------------------
// 8) bf16 GEMM with V_WMMA_F32_16X16X32_BF16 (fp32 accum), TDM-staged B.
//    Block = 4 waves; B slab (16 cols x KSLAB) DMA'd to LDS by the Tensor
//    Data Mover, double-buffered on TENSORcnt, shared by all waves.
//    Each wave computes a 32(M) x 16(N) C block (2 accumulators) so each
//    LDS B fragment feeds two WMMAs.
// 16-bit A frag layout (ISA 7.12.2): lane m=l&15, hi=l>>4; vgpr j holds the
// packed pair at dword offset {0..3,8..11}+4*hi within the 32-wide K step.
// B assumed symmetric (N per lane), fragments read from LDS.
// ---------------------------------------------------------------------------
template <int OUT_BF16>
__global__ __launch_bounds__(128)
void gemm_bf16_wmma(const uint16_t* __restrict__ A, int lda,
                    const uint16_t* __restrict__ Bt, int ldb,
                    const float* __restrict__ bias, int relu,
                    void* __restrict__ Cout, int ldc, int M, int N, int K)
{
  __shared__ uint16_t sB[2][16 * KSLAB];          // 2 x 4KB double buffer
  const int tid  = threadIdx.x;
  const int wave = tid >> 5;
  const int lane = tid & 31;
  const int lo = lane & 15, hi = lane >> 4;
  const int n0 = blockIdx.x * 16;
  const int m0 = blockIdx.y * 128 + wave * 32;    // 2 M-tiles per wave

  int mA0 = m0 + lo;       if (mA0 >= M) mA0 = 0; // clamp (stores guarded)
  int mA1 = m0 + 16 + lo;  if (mA1 >= M) mA1 = 0;
  const uint16_t* arow0 = A + (size_t)mA0 * lda;
  const uint16_t* arow1 = A + (size_t)mA1 * lda;
  const uint16_t* brow  = Bt + (size_t)n0 * ldb;  // slab source (16 rows)

  const int nslab = (K + KSLAB - 1) / KSLAB;
  if (wave == 0) {
    tdm_load_2d((uint32_t)(uintptr_t)&sB[0][0], brow, imin(KSLAB, K), 16, ldb);
  }

  v8f acc0 = {0.f,0.f,0.f,0.f,0.f,0.f,0.f,0.f};
  v8f acc1 = {0.f,0.f,0.f,0.f,0.f,0.f,0.f,0.f};

  for (int s = 0; s < nslab; ++s) {
    const int k0  = s * KSLAB;
    const int len = imin(KSLAB, K - k0);
    if (wave == 0) {
      if (s + 1 < nslab) {
        const int k1 = k0 + KSLAB;
        tdm_load_2d((uint32_t)(uintptr_t)&sB[(s + 1) & 1][0],
                    brow + k1, imin(KSLAB, K - k1), 16, ldb);
        __builtin_amdgcn_s_wait_tensorcnt((short)1);  // slab s complete
      } else {
        __builtin_amdgcn_s_wait_tensorcnt((short)0);
      }
    }
    __syncthreads();                                  // publish slab s
    const uint16_t* sbuf = &sB[s & 1][0];             // packed pitch = len
    for (int kk = 0; kk < len; kk += 32) {
      const uint32_t* ap0 = (const uint32_t*)(arow0 + k0 + kk);
      const uint32_t* ap1 = (const uint32_t*)(arow1 + k0 + kk);
      const uint32_t* bp  = (const uint32_t*)(sbuf + (size_t)lo * len + kk);
      union { v16bf v; uint32_t u[8]; } av0, av1, bv;
#pragma unroll
      for (int j = 0; j < 8; ++j) {
        const int off = j + ((j >> 2) << 2) + (hi << 2);  // {0..3,8..11}+4*hi
        av0.u[j] = ap0[off];
        av1.u[j] = ap1[off];
        bv.u[j]  = bp[off];
      }
      acc0 = __builtin_amdgcn_wmma_f32_16x16x32_bf16(false, av0.v, false, bv.v,
                                                     (short)0, acc0, false, false);
      acc1 = __builtin_amdgcn_wmma_f32_16x16x32_bf16(false, av1.v, false, bv.v,
                                                     (short)0, acc1, false, false);
    }
    __syncthreads();                                  // done reading slab s
  }

  const int nOut = n0 + lo;
  const float bb = bias ? bias[nOut] : 0.0f;
#pragma unroll
  for (int j = 0; j < 8; ++j) {
    const int m = m0 + hi * 8 + j;
    if (m < M) {
      float v = acc0[j] + bb;
      if (relu) v = v > 0.0f ? v : 0.0f;
      if (OUT_BF16) ((uint16_t*)Cout)[(size_t)m * ldc + nOut] = f2bf(v);
      else          ((float*)Cout)[(size_t)m * ldc + nOut]    = v;
    }
    const int m2 = m + 16;
    if (m2 < M) {
      float v = acc1[j] + bb;
      if (relu) v = v > 0.0f ? v : 0.0f;
      if (OUT_BF16) ((uint16_t*)Cout)[(size_t)m2 * ldc + nOut] = f2bf(v);
      else          ((float*)Cout)[(size_t)m2 * ldc + nOut]    = v;
    }
  }
}

// ---------------------------------------------------------------------------
// 9) final cls (5) + bbox (20) heads, concatenated -> (B,500,25)
// ---------------------------------------------------------------------------
__global__ void final_head_kernel(const float* __restrict__ x7,
                                  const float* __restrict__ cls_w, const float* __restrict__ cls_b,
                                  const float* __restrict__ bbox_w, const float* __restrict__ bbox_b,
                                  float* __restrict__ out)
{
  const int gid = blockIdx.x * blockDim.x + threadIdx.x;
  const int TOT = BATCH * POST_K * 25;
  if (gid >= TOT) return;
  const int m = gid / 25, j = gid % 25;
  const float* X = x7 + (size_t)m * HID;
  float s; const float* Wc; int ld;
  if (j < 5) { s = cls_b[j];      Wc = cls_w  + j;       ld = 5;  }
  else       { s = bbox_b[j - 5]; Wc = bbox_w + (j - 5); ld = 20; }
  for (int k = 0; k < HID; ++k) s += X[k] * Wc[(size_t)k * ld];
  out[(size_t)m * 25 + j] = s;
}

// ---------------------------------------------------------------------------
// Launch
// ---------------------------------------------------------------------------
static inline void* bump(char*& p, size_t bytes) {
  void* r = (void*)p;
  p += (bytes + 255) & ~(size_t)255;
  return r;
}

extern "C" void kernel_launch(void* const* d_in, const int* in_sizes, int n_in,
                              void* d_out, int out_size, void* d_ws, size_t ws_size,
                              hipStream_t stream) {
  (void)in_sizes; (void)n_in; (void)out_size; (void)ws_size;
  const float* features   = (const float*)d_in[0];
  const float* rpn_conv_w = (const float*)d_in[1];
  const float* rpn_conv_b = (const float*)d_in[2];
  const float* rpn_cls_w  = (const float*)d_in[3];
  const float* rpn_cls_b  = (const float*)d_in[4];
  const float* rpn_reg_w  = (const float*)d_in[5];
  const float* rpn_reg_b  = (const float*)d_in[6];
  const float* fc6_w      = (const float*)d_in[7];
  const float* fc6_b      = (const float*)d_in[8];
  const float* fc7_w      = (const float*)d_in[9];
  const float* fc7_b      = (const float*)d_in[10];
  const float* cls_w      = (const float*)d_in[11];
  const float* cls_b      = (const float*)d_in[12];
  const float* bbox_w     = (const float*)d_in[13];
  const float* bbox_b     = (const float*)d_in[14];

  char* p = (char*)d_ws;
  float*    t      = (float*)   bump(p, (size_t)BATCH * CIN * NPOS * 4);
  float*    obj    = (float*)   bump(p, (size_t)BATCH * NTOT * 4);
  float*    reg    = (float*)   bump(p, (size_t)BATCH * NTOT * 4 * 4);
  float*    sel_s  = (float*)   bump(p, (size_t)BATCH * PRE_K * 4);
  int*      sel_i  = (int*)     bump(p, (size_t)BATCH * PRE_K * 4);
  float*    boxes  = (float*)   bump(p, (size_t)BATCH * PRE_K * 4 * 4);
  float*    props  = (float*)   bump(p, (size_t)BATCH * POST_K * 4 * 4);
  uint16_t* pooled = (uint16_t*)bump(p, (size_t)BATCH * POST_K * FCK * 2);
  uint16_t* Bt6    = (uint16_t*)bump(p, (size_t)HID * FCK * 2);
  uint16_t* Bt7    = (uint16_t*)bump(p, (size_t)HID * HID * 2);
  uint16_t* x6     = (uint16_t*)bump(p, (size_t)BATCH * POST_K * HID * 2);
  float*    x7     = (float*)   bump(p, (size_t)BATCH * POST_K * HID * 4);

  const int M = BATCH * POST_K;  // 1000

  // weight repack (fp32 (K,N) -> bf16 (N,K))
  {
    const size_t e6 = (size_t)FCK * HID;
    repack_w_bf16<<<dim3((unsigned)((e6 + 255) / 256)), 256, 0, stream>>>(fc6_w, Bt6, FCK, HID);
    const size_t e7 = (size_t)HID * HID;
    repack_w_bf16<<<dim3((unsigned)((e7 + 255) / 256)), 256, 0, stream>>>(fc7_w, Bt7, HID, HID);
  }

  // RPN conv (fp32 WMMA implicit GEMM) + heads
  rpn_conv_wmma<<<dim3(CIN / 16, (BATCH * NPOS) / 16), 32, 0, stream>>>(
      features, rpn_conv_w, rpn_conv_b, t);
  rpn_head_kernel<<<(BATCH * NPOS * 60 + 255) / 256, 256, 0, stream>>>(
      t, rpn_cls_w, rpn_cls_b, rpn_reg_w, rpn_reg_b, obj, reg);

  // proposal selection
  topk_obj_kernel<<<dim3((NTOT + 255) / 256, BATCH), 256, 0, stream>>>(obj, sel_s, sel_i);
  decode_kernel<<<(BATCH * PRE_K + 255) / 256, 256, 0, stream>>>(sel_i, reg, boxes);
  nms_select_kernel<<<BATCH, 1024, 0, stream>>>(boxes, sel_s, props);

  // ROI-align -> bf16 pooled activations
  roi_align_kernel<<<dim3((CIN + 255) / 256, POST_K * POOLSZ * POOLSZ, BATCH),
                     256, 0, stream>>>(features, props, pooled);

  // fc6 / fc7 (bf16 WMMA GEMMs with TDM-staged, double-buffered B)
  gemm_bf16_wmma<1><<<dim3(HID / 16, (M + 127) / 128), 128, 0, stream>>>(
      pooled, FCK, Bt6, FCK, fc6_b, 1, (void*)x6, HID, M, HID, FCK);
  gemm_bf16_wmma<0><<<dim3(HID / 16, (M + 127) / 128), 128, 0, stream>>>(
      x6, HID, Bt7, HID, fc7_b, 1, (void*)x7, HID, M, HID, HID);

  // final heads -> output
  final_head_kernel<<<(BATCH * POST_K * 25 + 255) / 256, 256, 0, stream>>>(
      x7, cls_w, cls_b, bbox_w, bbox_b, (float*)d_out);
}
